// CoreAttention_25812753449354
// MI455X (gfx1250) — matrix operand order repaired and checked by reading, not compiled, not run on hardware
//
#include <hip/hip_runtime.h>

typedef __attribute__((ext_vector_type(16))) _Float16 v16h;
typedef __attribute__((ext_vector_type(8)))  float    v8f;

#define SQ 2048
#define BB 2
#define NHH 16
#define HN 128
#define ROW_STRIDE (BB*NHH*HN)   /* 4096 floats between consecutive s */
#define OUT_ROW (NHH*HN)         /* 2048 */

__launch_bounds__(128, 1)
__global__ void fa_fwd_causal(const float* __restrict__ Q,
                              const float* __restrict__ K,
                              const float* __restrict__ V,
                              float* __restrict__ Out)
{
    // per-wave, per-M-tile 16x32 f16 P staging tiles (4 waves x 2 tiles x 1KB = 8KB)
    __shared__ __align__(16) _Float16 sP[4][2][16 * 32];

    const int tid  = threadIdx.x;
    const int wid  = tid >> 5;
    const int lane = tid & 31;
    const int half = lane >> 4;     // 0: lanes 0-15, 1: lanes 16-31
    const int ln   = lane & 15;

    const int q0w = (blockIdx.x * 4 + wid) * 32;    // first query row of this wave (32 rows)
    const int bh = blockIdx.y;
    const int b  = bh / NHH;
    const int h  = bh % NHH;
    const int base_bh = (b * NHH + h) * HN;

    const float scale = 0.08838834764831845f;       // 1/sqrt(128)

    // ---------------- Q tiles: 2 M-tiles x 4 A-fragments (16x32 f16), K-dim = feature ----
    // A layout: M = lane&15 ; element e -> k = (e>=8 ? 16 : 0) + half*8 + (e&7)
    v16h aQ[2][4];
    #pragma unroll
    for (int u = 0; u < 2; ++u) {
        const float* qrow = Q + (size_t)(q0w + u * 16 + ln) * ROW_STRIDE + base_bh;
        #pragma unroll
        for (int c = 0; c < 4; ++c) {
            #pragma unroll
            for (int g = 0; g < 2; ++g) {
                const float* p = qrow + c * 32 + g * 16 + half * 8;
                float4 f0 = *(const float4*)(p);
                float4 f1 = *(const float4*)(p + 4);
                aQ[u][c][g * 8 + 0] = (_Float16)f0.x;
                aQ[u][c][g * 8 + 1] = (_Float16)f0.y;
                aQ[u][c][g * 8 + 2] = (_Float16)f0.z;
                aQ[u][c][g * 8 + 3] = (_Float16)f0.w;
                aQ[u][c][g * 8 + 4] = (_Float16)f1.x;
                aQ[u][c][g * 8 + 5] = (_Float16)f1.y;
                aQ[u][c][g * 8 + 6] = (_Float16)f1.z;
                aQ[u][c][g * 8 + 7] = (_Float16)f1.w;
            }
        }
    }

    // ---------------- output accumulators + online-softmax state ----------------
    v8f accO[2][8];
    #pragma unroll
    for (int u = 0; u < 2; ++u)
        #pragma unroll
        for (int f = 0; f < 8; ++f)
            #pragma unroll
            for (int i = 0; i < 8; ++i) accO[u][f][i] = 0.0f;

    float mstate[2][8], lstate[2][8];
    #pragma unroll
    for (int u = 0; u < 2; ++u)
        #pragma unroll
        for (int r = 0; r < 8; ++r) { mstate[u][r] = -1.0e30f; lstate[u][r] = 0.0f; }

    const int nkb = ((q0w + 31) >> 5) + 1;          // causal: 32-key blocks

    for (int kb = 0; kb < nkb; ++kb) {
        const int jb = kb * 32;
        float s[2][2][8];                           // [m-tile u][key-tile t][row r]

        // ------- S = Q * K^T : per key-tile, one K fragment feeds BOTH M-tiles -------
        #pragma unroll
        for (int t = 0; t < 2; ++t) {
            v8f accS[2];
            #pragma unroll
            for (int u = 0; u < 2; ++u)
                #pragma unroll
                for (int i = 0; i < 8; ++i) accS[u][i] = 0.0f;

            // B layout: N(key) = lane&15 ; element e -> k(feature in chunk) = (e>=8 ? 16 : 0) + half*8 + (e&7)
            const float* krow = K + (size_t)(jb + t * 16 + ln) * ROW_STRIDE + base_bh;
            #pragma unroll
            for (int c = 0; c < 4; ++c) {
                const float* p = krow + c * 32 + half * 8;
                float4 f0 = *(const float4*)(p);
                float4 f1 = *(const float4*)(p + 4);
                float4 f2 = *(const float4*)(p + 16);
                float4 f3 = *(const float4*)(p + 20);
                v16h bk;
                bk[0]  = (_Float16)f0.x; bk[1]  = (_Float16)f0.y;
                bk[2]  = (_Float16)f0.z; bk[3]  = (_Float16)f0.w;
                bk[4]  = (_Float16)f1.x; bk[5]  = (_Float16)f1.y;
                bk[6]  = (_Float16)f1.z; bk[7]  = (_Float16)f1.w;
                bk[8]  = (_Float16)f2.x; bk[9]  = (_Float16)f2.y;
                bk[10] = (_Float16)f2.z; bk[11] = (_Float16)f2.w;
                bk[12] = (_Float16)f3.x; bk[13] = (_Float16)f3.y;
                bk[14] = (_Float16)f3.z; bk[15] = (_Float16)f3.w;
                #pragma unroll
                for (int u = 0; u < 2; ++u) {
                    accS[u] = __builtin_amdgcn_wmma_f32_16x16x32_f16(
                                false, aQ[u][c], false, bk, (short)0, accS[u], false, false);
                }
            }

            // scale + causal mask. C/D layout: N = lane&15, M = r + 8*half
            const int kg = jb + t * 16 + ln;
            #pragma unroll
            for (int u = 0; u < 2; ++u) {
                #pragma unroll
                for (int r = 0; r < 8; ++r) {
                    const int qg = q0w + u * 16 + r + half * 8;
                    float v = accS[u][r] * scale;
                    s[u][t][r] = (kg > qg) ? -10000.0f : v;
                }
            }
        }

        // ------- online softmax per M-tile / row, stage P into LDS -------
        v16h aP[2];
        #pragma unroll
        for (int u = 0; u < 2; ++u) {
            #pragma unroll
            for (int r = 0; r < 8; ++r) {
                float mx = fmaxf(s[u][0][r], s[u][1][r]);
                mx = fmaxf(mx, __shfl_xor(mx, 1, 32));
                mx = fmaxf(mx, __shfl_xor(mx, 2, 32));
                mx = fmaxf(mx, __shfl_xor(mx, 4, 32));
                mx = fmaxf(mx, __shfl_xor(mx, 8, 32));
                const float mo = mstate[u][r];
                const float mn = fmaxf(mo, mx);
                const float p0 = __expf(s[u][0][r] - mn);
                const float p1 = __expf(s[u][1][r] - mn);
                float sum = p0 + p1;
                sum += __shfl_xor(sum, 1, 32);
                sum += __shfl_xor(sum, 2, 32);
                sum += __shfl_xor(sum, 4, 32);
                sum += __shfl_xor(sum, 8, 32);
                const float alpha = __expf(mo - mn);
                lstate[u][r] = lstate[u][r] * alpha + sum;
                mstate[u][r] = mn;
                #pragma unroll
                for (int f = 0; f < 8; ++f) accO[u][f][r] *= alpha;
                // D-layout -> LDS at [row m][key k] (f16)
                sP[wid][u][(r + half * 8) * 32 + ln]      = (_Float16)p0;
                sP[wid][u][(r + half * 8) * 32 + 16 + ln] = (_Float16)p1;
            }
            // reload P in A layout (16x32, K-dim = key); same-wave DS ordering
            const _Float16* pb = &sP[wid][u][ln * 32 + half * 8];
            #pragma unroll
            for (int e = 0; e < 8; ++e) {
                aP[u][e]     = pb[e];          // k = half*8 + e
                aP[u][8 + e] = pb[16 + e];     // k = 16 + half*8 + e
            }
        }

        // ------- O += P * V : one V fragment feeds BOTH M-tiles -------
        #pragma unroll
        for (int f = 0; f < 8; ++f) {
            v16h bv;   // B layout: N(feature) = lane&15, element e -> key = (e>=8 ? 16 : 0) + half*8 + (e&7)
            const float* vcol = V + base_bh + f * 16 + ln;
            #pragma unroll
            for (int e = 0; e < 16; ++e) {
                bv[e] = (_Float16)vcol[(size_t)(jb + (e >> 3) * 16 + half * 8 + (e & 7)) * ROW_STRIDE];
            }
            #pragma unroll
            for (int u = 0; u < 2; ++u) {
                accO[u][f] = __builtin_amdgcn_wmma_f32_16x16x32_f16(
                                false, aP[u], false, bv, (short)0, accO[u][f], false, false);
            }
        }

        // prefetch next key block (emits global_prefetch_b8)
        if (kb + 1 < nkb) {
            __builtin_prefetch(K + (size_t)(jb + 32 + ln) * ROW_STRIDE + base_bh, 0, 1);
            __builtin_prefetch(V + (size_t)(jb + 32 + ln) * ROW_STRIDE + base_bh, 0, 1);
        }
    }

    // ---------------- epilogue: normalize and store fp32 ----------------
    #pragma unroll
    for (int u = 0; u < 2; ++u) {
        #pragma unroll
        for (int r = 0; r < 8; ++r) {
            const int qg = q0w + u * 16 + r + half * 8;
            const float inv = 1.0f / lstate[u][r];
            float* orow = Out + (size_t)(qg * BB + b) * OUT_ROW + h * HN;
            #pragma unroll
            for (int f = 0; f < 8; ++f) {
                orow[f * 16 + ln] = accO[u][f][r] * inv;
            }
        }
    }
}

extern "C" void kernel_launch(void* const* d_in, const int* in_sizes, int n_in,
                              void* d_out, int out_size, void* d_ws, size_t ws_size,
                              hipStream_t stream) {
    (void)in_sizes; (void)n_in; (void)out_size; (void)d_ws; (void)ws_size;
    const float* Q = (const float*)d_in[0];
    const float* K = (const float*)d_in[1];
    const float* V = (const float*)d_in[2];
    // d_in[3] = attention_mask (bool causal) — implemented arithmetically in-kernel
    float* out = (float*)d_out;

    dim3 grid(SQ / 128, BB * NHH);   // 16 x 32
    dim3 block(128);                 // 4 waves x 32 query rows
    fa_fwd_causal<<<grid, block, 0, stream>>>(Q, K, V, out);
}
